// CALayer_62027917689458
// MI455X (gfx1250) — compile-verified, matrix operand order checked by reading, not run on hardware
//
#include <hip/hip_runtime.h>
#include <math.h>

// ---------------------------------------------------------------------------
// CALayer (squeeze-excite + top-k channel gather) for MI455X / gfx1250.
// Memory-bound problem: ~192 MB effective HBM traffic => ~8 us @ 23.3 TB/s.
// WMMA (v_wmma_f32_16x16x32_f16) used for the SE-MLP; NT stores keep the
// 128 MB input resident in the 192 MB L2 for the gather pass.
// ---------------------------------------------------------------------------

#define B_    32
#define C_    256
#define RR    16      // C / reduction
#define HW_   4096    // 64*64
#define TOPK_ 128

typedef __attribute__((ext_vector_type(16))) _Float16 v16h;
typedef __attribute__((ext_vector_type(8)))  float    v8f;
typedef __attribute__((ext_vector_type(4)))  float    f4;

union AV { v16h v; _Float16 e[16]; };

// ---------------------------------------------------------------------------
// Kernel 1: global average pool.  One block per (b,c) plane of 4096 floats.
// 128 MB streamed from HBM with b128 loads (fills L2 with x: regular TH).
// ---------------------------------------------------------------------------
__global__ __launch_bounds__(256)
void pool_kernel(const float* __restrict__ x, float* __restrict__ y) {
    __shared__ float red[256];
    const int t = threadIdx.x;
    const f4* p = (const f4*)(x + (size_t)blockIdx.x * HW_);
    float s = 0.0f;
#pragma unroll
    for (int i = 0; i < 4; ++i) {
        f4 v = p[t + 256 * i];                 // global_load_b128
        s += v[0] + v[1] + v[2] + v[3];
    }
    red[t] = s;
    __syncthreads();
    for (int off = 128; off > 0; off >>= 1) {
        if (t < off) red[t] += red[t + off];
        __syncthreads();
    }
    if (t == 0) y[blockIdx.x] = red[0] * (1.0f / (float)HW_);
}

// ---------------------------------------------------------------------------
// Kernel 2: SE-MLP with WMMA.  Single wave32 (EXEC all ones, as required).
//   h    = relu(y[32x256] @ w1^T + b1)        -> 2 M-tiles x 8 K-steps
//   attn = sigmoid(h[32x16] @ w2^T + b2)      -> 2 M-tiles x 16 N-tiles (K pad)
// A/B per-lane layouts per ISA 7.12.2 (16-bit A 16x32; B assumed symmetric).
// ---------------------------------------------------------------------------
__global__ __launch_bounds__(32)
void mlp_wmma_kernel(const float* __restrict__ y,
                     const float* __restrict__ w1, const float* __restrict__ b1,
                     const float* __restrict__ w2, const float* __restrict__ b2,
                     float* __restrict__ attn) {
    __shared__ float hbuf[B_][RR];
    const int lane = threadIdx.x;          // 0..31
    const int hs   = (lane < 16) ? 0 : 8;  // K offset for upper lane half
    const int lm   = lane & 15;            // local M (A) / local N (B,D)

    // ---- layer 1: h = relu(y @ w1^T + b1), h[m][r] ----
    for (int mt = 0; mt < 2; ++mt) {
        v8f acc = {};
        for (int kt = 0; kt < 8; ++kt) {
            AV a, b;
#pragma unroll
            for (int i = 0; i < 16; ++i) {
                const int v = i >> 1, j = i & 1;
                const int k = kt * 32 + ((v < 4) ? 0 : 16) + hs + 2 * (v & 3) + j;
                a.e[i] = (_Float16)y[(mt * 16 + lm) * C_ + k];   // A[m][k]
                b.e[i] = (_Float16)w1[lm * C_ + k];              // B[k][n]=w1[n][k]
            }
            acc = __builtin_amdgcn_wmma_f32_16x16x32_f16(
                      false, a.v, false, b.v, (short)0, acc, false, false);
        }
#pragma unroll
        for (int r = 0; r < 8; ++r) {
            const int m = mt * 16 + r + hs;      // C/D layout: VGPR r, lane half
            const float hv = acc[r] + b1[lm];
            hbuf[m][lm] = hv > 0.0f ? hv : 0.0f;
        }
    }
    __syncthreads();

    // ---- layer 2: attn = sigmoid(h @ w2^T + b2), K=16 zero-padded to 32 ----
    for (int mt = 0; mt < 2; ++mt) {
        AV a;
#pragma unroll
        for (int i = 0; i < 16; ++i) {
            const int v = i >> 1, j = i & 1;
            const int k = ((v < 4) ? 0 : 16) + hs + 2 * (v & 3) + j;
            a.e[i] = (k < RR) ? (_Float16)hbuf[mt * 16 + lm][k] : (_Float16)0.0f;
        }
        for (int nt = 0; nt < 16; ++nt) {
            AV b;
#pragma unroll
            for (int i = 0; i < 16; ++i) {
                const int v = i >> 1, j = i & 1;
                const int k = ((v < 4) ? 0 : 16) + hs + 2 * (v & 3) + j;
                b.e[i] = (k < RR) ? (_Float16)w2[(nt * 16 + lm) * RR + k]
                                  : (_Float16)0.0f;                 // B[k][n]=w2[n][k]
            }
            v8f acc = {};
            acc = __builtin_amdgcn_wmma_f32_16x16x32_f16(
                      false, a.v, false, b.v, (short)0, acc, false, false);
#pragma unroll
            for (int r = 0; r < 8; ++r) {
                const int m = mt * 16 + r + hs;
                const int c = nt * 16 + lm;
                const float z = acc[r] + b2[c];
                attn[m * C_ + c] = 1.0f / (1.0f + __expf(-z));
            }
        }
    }
}

// ---------------------------------------------------------------------------
// Kernel 3: per-batch top-128 of 256 via rank counting (jax tie-break: lower
// index wins), compacted in ascending channel order. One block per batch.
// ---------------------------------------------------------------------------
__global__ __launch_bounds__(256)
void topk_kernel(const float* __restrict__ attn,
                 int* __restrict__ idx, float* __restrict__ aw) {
    __shared__ float a[C_];
    __shared__ int   sel[C_];
    const int b = blockIdx.x, c = threadIdx.x;
    a[c] = attn[b * C_ + c];
    __syncthreads();
    const float me = a[c];
    int rank = 0;
    for (int c2 = 0; c2 < C_; ++c2) {
        const float o = a[c2];
        rank += (o > me) || (o == me && c2 < c);
    }
    const int s = (rank < TOPK_) ? 1 : 0;
    sel[c] = s;
    __syncthreads();
    if (s) {
        int pos = 0;
        for (int c2 = 0; c2 < c; ++c2) pos += sel[c2];
        idx[b * TOPK_ + pos] = c;
        aw[b * TOPK_ + pos]  = me;
    }
}

// ---------------------------------------------------------------------------
// Kernel 4: gather selected channels and scale.  x should still be L2-resident
// (128 MB < 192 MB L2); output written with non-temporal stores so the 64 MB
// of output does not evict x.
// ---------------------------------------------------------------------------
__global__ __launch_bounds__(256)
void gather_scale_kernel(const float* __restrict__ x,
                         const int* __restrict__ idx,
                         const float* __restrict__ aw,
                         float* __restrict__ out) {
    const int bj = blockIdx.x;             // b*128 + j
    const int b  = bj >> 7;
    const int c  = idx[bj];
    const float w = aw[bj];
    const f4* src = (const f4*)(x + (size_t)(b * C_ + c) * HW_);
    f4*       dst = (f4*)(out + (size_t)bj * HW_);
    const int t = threadIdx.x;
#pragma unroll
    for (int i = 0; i < 4; ++i) {
        f4 v = src[t + 256 * i];           // global_load_b128 (L2 hit)
        f4 r = v * w;
        __builtin_nontemporal_store(r, &dst[t + 256 * i]);  // NT store_b128
    }
}

// ---------------------------------------------------------------------------
extern "C" void kernel_launch(void* const* d_in, const int* in_sizes, int n_in,
                              void* d_out, int out_size, void* d_ws, size_t ws_size,
                              hipStream_t stream) {
    const float* x  = (const float*)d_in[0];
    const float* w1 = (const float*)d_in[1];
    const float* b1 = (const float*)d_in[2];
    const float* w2 = (const float*)d_in[3];
    const float* b2 = (const float*)d_in[4];
    // d_in[5] = topk scalar (== 128, compile-time constant here)
    float* out = (float*)d_out;

    char*  ws   = (char*)d_ws;
    float* y    = (float*)(ws);               //  8192 f32 (32 KB)
    float* attn = (float*)(ws + 32 * 1024);   //  8192 f32 (32 KB)
    int*   idx  = (int*)  (ws + 64 * 1024);   //  4096 i32 (16 KB)
    float* aw   = (float*)(ws + 80 * 1024);   //  4096 f32 (16 KB)

    pool_kernel<<<B_ * C_, 256, 0, stream>>>(x, y);
    mlp_wmma_kernel<<<1, 32, 0, stream>>>(y, w1, b1, w2, b2, attn);
    topk_kernel<<<B_, 256, 0, stream>>>(attn, idx, aw);
    gather_scale_kernel<<<B_ * TOPK_, 256, 0, stream>>>(x, idx, aw, out);
}
